// GATLayer_17154099380254
// MI455X (gfx1250) — matrix operand order repaired and checked by reading, not compiled, not run on hardware
//
#include <hip/hip_runtime.h>

typedef float    v2f  __attribute__((ext_vector_type(2)));
typedef float    v8f  __attribute__((ext_vector_type(8)));
typedef _Float16 v16h __attribute__((ext_vector_type(16)));
typedef _Float16 v8h  __attribute__((ext_vector_type(8)));
typedef unsigned int v4u __attribute__((ext_vector_type(4)));
typedef int      v4i  __attribute__((ext_vector_type(4)));
typedef int      v8i  __attribute__((ext_vector_type(8)));

#define N_NODES 4096
#define D_IN    512
#define D_OUT   128
#define NHEAD   4
#define NT      (N_NODES / 32)   // 128 j-tiles in kernel 3

// ---------------------------------------------------------------------------
// TDM helper: issue a 2D tensor_load_to_lds (D# per CDNA5 ISA).
// All dims/strides in data_size units. Optional LDS row padding.
//   pad_interval_code: pad after 2<<code dwords;  pad_amount_code: (code+1) dwords
// Issued once per calling wave (EXEC ignored by TDM) -> guard at wave level.
// ---------------------------------------------------------------------------
__device__ __forceinline__ void tdm_load_2d(unsigned int lds_addr, const void* gptr,
                                            unsigned int data_size_code,
                                            unsigned int tensor_d0, unsigned int tensor_d1,
                                            unsigned int tile_d0, unsigned int tile_d1,
                                            unsigned int d0_stride,
                                            unsigned int pad_interval_code,
                                            unsigned int pad_amount_code) {
    const unsigned long long ga = (unsigned long long)(uintptr_t)gptr;
    v4u g0;
    g0.x = 1u;                                   // count=1, user descriptor
    g0.y = lds_addr;                             // LDS byte address
    g0.z = (unsigned int)ga;                     // global_addr[31:0]
    g0.w = (unsigned int)((ga >> 32) & 0x1FFFFFFu) | (2u << 30);  // addr[56:32] | type=2
    v8i g1;
    g1[0] = (int)((data_size_code << 16) | (1u << 20) |
                  (pad_interval_code << 22) | (pad_amount_code << 25)); // pad_enable=1
    g1[1] = (int)((tensor_d0 & 0xFFFFu) << 16);
    g1[2] = (int)(((tensor_d0 >> 16) & 0xFFFFu) | ((tensor_d1 & 0xFFFFu) << 16));
    g1[3] = (int)(((tensor_d1 >> 16) & 0xFFFFu) | (tile_d0 << 16));
    g1[4] = (int)tile_d1;                        // tile_dim1; tile_dim2=0 (2D)
    g1[5] = (int)d0_stride;                      // tensor_dim0_stride[31:0]
    g1[6] = 0;                                   // stride hi | dim1_stride lo
    g1[7] = 0;
    v4i gz = {0, 0, 0, 0};
#if __clang_major__ >= 23
    v8i gz8 = {0, 0, 0, 0, 0, 0, 0, 0};
    __builtin_amdgcn_tensor_load_to_lds(g0, g1, gz, gz, gz8, 0);
#else
    __builtin_amdgcn_tensor_load_to_lds(g0, g1, gz, gz, 0);
#endif
}

__device__ __forceinline__ unsigned int lds_offset(const void* p) {
    return (unsigned int)(uintptr_t)p;           // LDS aperture: offset = addr[31:0]
}

// ---------------------------------------------------------------------------
// Kernel 1: h = x @ W  (fp32, exact, V_WMMA_F32_16X16X4_F32)
// Also emits hfT = f16 transpose of h, [D_OUT][N], for kernel 3's B-operand.
// ---------------------------------------------------------------------------
__global__ __launch_bounds__(256)
void gemm_h_kernel(const float* __restrict__ x, const float* __restrict__ W,
                   float* __restrict__ h, _Float16* __restrict__ hfT) {
    __shared__ __attribute__((aligned(16))) float xs[16][36];  // 36: 16B-aligned rows, conflict-free
    const int tid   = threadIdx.x;
    const int lane  = tid & 31;
    const int wave  = tid >> 5;
    const int iBase = blockIdx.x * 16;
    const int nBase = wave * 16;
    const int row   = lane & 15;
    const int kb    = (lane >> 4) * 2;   // f32 A/B layout: K{0,1} lanes 0-15, K{2,3} lanes 16-31

    v8f c;
    for (int r = 0; r < 8; ++r) c[r] = 0.0f;

    for (int k0 = 0; k0 < D_IN; k0 += 32) {
        if (tid < 128) {                 // one float4 per thread: 16x32 tile
            const int r  = tid >> 3;
            const int cc = (tid & 7) * 4;
            *(float4*)&xs[r][cc] =
                *(const float4*)&x[(size_t)(iBase + r) * D_IN + k0 + cc];
        }
        __syncthreads();
#pragma unroll
        for (int kk = 0; kk < 32; kk += 4) {
            v2f a, b;
            a.x = xs[row][kk + kb];
            a.y = xs[row][kk + kb + 1];
            const int gk = k0 + kk + kb;
            b.x = W[(size_t)gk       * D_OUT + nBase + row];
            b.y = W[(size_t)(gk + 1) * D_OUT + nBase + row];
            c = __builtin_amdgcn_wmma_f32_16x16x4_f32(false, a, false, b,
                                                      (short)0, c, false, false);
        }
        __syncthreads();
    }

    const int base8 = (lane >> 4) * 8;   // C/D: M = r + (lane>=16 ? 8 : 0)
    const int col   = lane & 15;
    v8h hv;
#pragma unroll
    for (int r = 0; r < 8; ++r) {
        const int M = r + base8;
        h[(size_t)(iBase + M) * D_OUT + nBase + col] = c[r];
        hv[r] = (_Float16)c[r];
    }
    *(uint4*)&hfT[(size_t)(nBase + col) * N_NODES + iBase + base8] = *(uint4*)&hv;
}

// ---------------------------------------------------------------------------
// Kernel 2: el/er = h @ attn_{l,r}^T  (tiny; one wave32 per node, shfl reduce)
// ---------------------------------------------------------------------------
__global__ __launch_bounds__(256)
void attn_proj_kernel(const float* __restrict__ h, const float* __restrict__ Al,
                      const float* __restrict__ Ar, float* __restrict__ el,
                      float* __restrict__ er) {
    const int lane = threadIdx.x & 31;
    const int node = blockIdx.x * 8 + (threadIdx.x >> 5);
    float hv[4];
#pragma unroll
    for (int t = 0; t < 4; ++t) hv[t] = h[(size_t)node * D_OUT + lane + 32 * t];
#pragma unroll
    for (int head = 0; head < NHEAD; ++head) {
        float sl = 0.f, sr = 0.f;
#pragma unroll
        for (int t = 0; t < 4; ++t) {
            sl += hv[t] * Al[head * D_OUT + lane + 32 * t];
            sr += hv[t] * Ar[head * D_OUT + lane + 32 * t];
        }
#pragma unroll
        for (int off = 16; off >= 1; off >>= 1) {
            sl += __shfl_xor(sl, off);
            sr += __shfl_xor(sr, off);
        }
        if (lane == 0) {
            el[node * NHEAD + head] = sl;
            er[node * NHEAD + head] = sr;
        }
    }
}

// ---------------------------------------------------------------------------
// Kernel 3: fused attention + normalize + aggregate, TDM double-buffered.
// Grid: 128 blocks = 32-row i-tiles; 8 waves: head=w&3, row-group=w>>2.
// Wave 0 drives two TDM 2D descriptors per j-tile (adj 32x32 f32 pad->33,
// hfT 128x32 f16 pad->40), pipelined one tile ahead (s_wait_tensorcnt 2).
// ---------------------------------------------------------------------------
__global__ __launch_bounds__(256)
void gat_aggregate_kernel(const float* __restrict__ adj,
                          const _Float16* __restrict__ hfT,
                          const float* __restrict__ el,
                          const float* __restrict__ er,
                          const float* __restrict__ bias,
                          float* __restrict__ out) {
    __shared__ __attribute__((aligned(16))) float    adjS[2][32][33];
    __shared__ __attribute__((aligned(16))) _Float16 hfS[2][128 * 40];
    __shared__ __attribute__((aligned(16))) float    erS[32 * 4];

    const int tid   = threadIdx.x;
    const int lane  = tid & 31;
    const int wave  = tid >> 5;
    const int head  = wave & 3;
    const int g     = wave >> 2;                 // row-group 0/1
    const int iBase = blockIdx.x * 32;
    const int mrow  = g * 16 + (lane & 15);      // local row 0..31
    const int lo    = (lane < 16) ? 0 : 8;       // f16 A-layout K offset
    const float elv = el[(size_t)(iBase + mrow) * NHEAD + head];

    v8f c[8];
    for (int ct = 0; ct < 8; ++ct)
        for (int r = 0; r < 8; ++r) c[ct][r] = 0.f;
    float dsum = 0.f;

    const unsigned int adjLds[2] = { lds_offset(&adjS[0][0][0]), lds_offset(&adjS[1][0][0]) };
    const unsigned int hfLds[2]  = { lds_offset(&hfS[0][0]),     lds_offset(&hfS[1][0]) };
    const float* adjRow = adj + (size_t)iBase * N_NODES;

    if (wave == 0) {   // prologue: tile 0 -> buffer 0
        tdm_load_2d(adjLds[0], adjRow, 2u, N_NODES, 32u, 32u, 32u, N_NODES, 4u, 0u);
        tdm_load_2d(hfLds[0],  hfT,    1u, N_NODES, 128u, 32u, 128u, N_NODES, 3u, 3u);
    }

    for (int t = 0; t < NT; ++t) {
        const int buf = t & 1;
        const int j0  = t * 32;

        if (wave == 0) {
            if (t + 1 < NT) {   // prefetch tile t+1 into the other buffer
                const int jn = j0 + 32;
                tdm_load_2d(adjLds[buf ^ 1], adjRow + jn, 2u, N_NODES, 32u, 32u, 32u, N_NODES, 4u, 0u);
                tdm_load_2d(hfLds[buf ^ 1],  hfT + jn,    1u, N_NODES, 128u, 32u, 128u, N_NODES, 3u, 3u);
                __builtin_amdgcn_s_wait_tensorcnt((short)2);   // tile t complete
            } else {
                __builtin_amdgcn_s_wait_tensorcnt((short)0);
            }
        }
        if (tid < 32)   // er tile [32 x 4] = 512B, one uint4 per thread
            *(uint4*)&erS[tid * 4] = *(const uint4*)&er[(size_t)(j0 + tid) * NHEAD];
        __syncthreads();

        // A fragment: lane<16 holds K{0-7,16-23}, lane>=16 holds K{8-15,24-31}
        v16h a;
#pragma unroll
        for (int e = 0; e < 16; ++e) {
            const int jloc = (e < 8 ? e : e + 8) + lo;
            const float s  = elv + erS[jloc * 4 + head];
            const float lr = fmaxf(s, 0.2f * s);                    // leaky_relu 0.2
            const float av = adjS[buf][mrow][jloc] * __expf(lr);    // adj is 0/1
            dsum += av;
            a[e] = (_Float16)av;
        }

        // B fragments from transposed LDS tile: lane<16 holds K{0-7,16-23}, lane>=16 holds K{8-15,24-31}
        const int baseK = (lane < 16) ? 0 : 8;
        const int ncol  = lane & 15;
#pragma unroll
        for (int ct = 0; ct < 8; ++ct) {
            v16h b;
            const uint4* p = (const uint4*)&hfS[buf][(ct * 16 + ncol) * 40 + baseK];
            ((uint4*)&b)[0] = p[0];
            ((uint4*)&b)[1] = p[2];
            c[ct] = __builtin_amdgcn_wmma_f32_16x16x32_f16(false, a, false, b,
                                                           (short)0, c[ct],
                                                           false, false);
        }
        __syncthreads();
    }

    // denom: lanes L and L^16 cover complementary K halves of row (L&15)
    dsum += __shfl_xor(dsum, 16);
    const float inv   = 1.0f / fmaxf(dsum, 1e-12f);
    const int   base8 = (lane >> 4) * 8;
    const int   ncol  = lane & 15;
#pragma unroll
    for (int ct = 0; ct < 8; ++ct) {
#pragma unroll
        for (int r = 0; r < 8; ++r) {
            const int   M    = r + base8;           // local row 0..15
            const float invr = __shfl(inv, M);      // lane M holds row M's inv
            const int   node = iBase + g * 16 + M;
            const int   col  = ct * 16 + ncol;
            out[(size_t)node * (NHEAD * D_OUT) + head * D_OUT + col] =
                c[ct][r] * invr + bias[col];
        }
    }
}

// ---------------------------------------------------------------------------
extern "C" void kernel_launch(void* const* d_in, const int* in_sizes, int n_in,
                              void* d_out, int out_size, void* d_ws, size_t ws_size,
                              hipStream_t stream) {
    const float* adj = (const float*)d_in[0];
    const float* x   = (const float*)d_in[1];
    const float* W   = (const float*)d_in[2];
    const float* Al  = (const float*)d_in[3];
    const float* Ar  = (const float*)d_in[4];
    const float* b   = (const float*)d_in[5];
    float* out = (float*)d_out;

    char* ws = (char*)d_ws;
    float*    h   = (float*)ws;                                      // 4096*128 f32
    _Float16* hfT = (_Float16*)(ws + (size_t)N_NODES * D_OUT * 4);   // 128*4096 f16
    float*    el  = (float*)(ws + (size_t)N_NODES * D_OUT * 4
                                + (size_t)D_OUT * N_NODES * 2);      // 4096*4
    float*    er  = el + (size_t)N_NODES * NHEAD;                    // 4096*4

    gemm_h_kernel<<<N_NODES / 16, 256, 0, stream>>>(x, W, h, hfT);
    attn_proj_kernel<<<N_NODES / 8, 256, 0, stream>>>(h, Al, Ar, el, er);
    gat_aggregate_kernel<<<N_NODES / 32, 256, 0, stream>>>(adj, hfT, el, er, b, out);
}
